// NFVLayer_18167711662349
// MI455X (gfx1250) — matrix operand order repaired and checked by reading, not compiled, not run on hardware
//
#include <hip/hip_runtime.h>
#include <hip/hip_bf16.h>

typedef __bf16 bf16_t;
typedef __attribute__((ext_vector_type(16))) __bf16 bf16x16;
typedef __attribute__((ext_vector_type(8)))  float  v8f;

#define IN_DIM 256
#define HID    64
#define HEADS  4
#define OUTF   256   // HEADS*HID
#define NEG_SLOPE 0.2f

// ---------------- W -> Wt (transposed, bf16): Wt[col][k] = W[k][col] ----------------
__global__ void k_wt(const float* __restrict__ W, bf16_t* __restrict__ Wt) {
    int idx = blockIdx.x * blockDim.x + threadIdx.x;
    if (idx >= IN_DIM * OUTF) return;
    int c = idx >> 8;     // output column 0..255
    int k = idx & 255;    // reduction index 0..255
    Wt[idx] = (bf16_t)W[k * OUTF + c];
}

// ---------------- zero helper ----------------
__global__ void k_zero(float* __restrict__ p, int n) {
    int i = blockIdx.x * blockDim.x + threadIdx.x;
    if (i < n) p[i] = 0.f;
}

__device__ __forceinline__ bf16x16 cvt_afrag(float4 A0, float4 A1, float4 A2, float4 A3) {
    bf16x16 a;
    a[0]  = (bf16_t)A0.x; a[1]  = (bf16_t)A0.y; a[2]  = (bf16_t)A0.z; a[3]  = (bf16_t)A0.w;
    a[4]  = (bf16_t)A1.x; a[5]  = (bf16_t)A1.y; a[6]  = (bf16_t)A1.z; a[7]  = (bf16_t)A1.w;
    a[8]  = (bf16_t)A2.x; a[9]  = (bf16_t)A2.y; a[10] = (bf16_t)A2.z; a[11] = (bf16_t)A2.w;
    a[12] = (bf16_t)A3.x; a[13] = (bf16_t)A3.y; a[14] = (bf16_t)A3.z; a[15] = (bf16_t)A3.w;
    return a;
}

// ---------------- xt = x @ W via v_wmma_f32_16x16x32_bf16 ----------------
// Block = 256 threads = 8 waves. Wave handles one 16-row M-tile x 128 columns
// (8 N-tiles), K fully unrolled with manual double-buffering of A and B.
__global__ void k_gemm(const float* __restrict__ x, const bf16_t* __restrict__ Wt,
                       float* __restrict__ xt, int Mtiles) {
    int lane = threadIdx.x & 31;
    int wv   = threadIdx.x >> 5;                  // 0..7
    int mt   = blockIdx.x * 4 + (wv >> 1);
    if (mt >= Mtiles) return;                     // wave-uniform: EXEC stays all-ones
    int cg   = wv & 1;                            // column group: 0 -> cols 0..127, 1 -> 128..255
    int row0 = mt * 16;
    int h    = lane >> 4;                         // lane half
    int m    = lane & 15;                         // M index (A) / N index (B,D)

    v8f acc[8];
    #pragma unroll
    for (int t = 0; t < 8; ++t) acc[t] = (v8f){};

    // A: row (row0+m); per 32-K window: frag 0..15 at +16h (contiguous)
    const float*  aptr = x + (size_t)(row0 + m) * IN_DIM;
    // B: single base pointer; tile t at byte offset t*16*IN_DIM*2, K-step kk at kk*32*2,
    //    so all 8 loads per K-step are immediate offsets -> one s_clause.
    const bf16_t* bptr = Wt + (size_t)(cg * 128 + m) * IN_DIM + 16 * h;

    // ---- prologue: K-step 0 ----
    float4 A0 = *(const float4*)(aptr + 16 * h + 0);
    float4 A1 = *(const float4*)(aptr + 16 * h + 4);
    float4 A2 = *(const float4*)(aptr + 16 * h + 8);
    float4 A3 = *(const float4*)(aptr + 16 * h + 12);
    bf16x16 B0[8];
    #pragma unroll
    for (int t = 0; t < 8; ++t)
        B0[t] = *(const bf16x16*)(bptr + t * 16 * IN_DIM);

    #pragma unroll
    for (int kk = 0; kk < 8; ++kk) {
        // ---- prefetch K-step kk+1 while computing kk ----
        float4 N0, N1, N2, N3;
        bf16x16 B1[8];
        if (kk < 7) {
            int nb = (kk + 1) * 32;
            N0 = *(const float4*)(aptr + nb + 16 * h + 0);
            N1 = *(const float4*)(aptr + nb + 16 * h + 4);
            N2 = *(const float4*)(aptr + nb + 16 * h + 8);
            N3 = *(const float4*)(aptr + nb + 16 * h + 12);
            #pragma unroll
            for (int t = 0; t < 8; ++t)
                B1[t] = *(const bf16x16*)(bptr + t * 16 * IN_DIM + nb);
        }

        bf16x16 a = cvt_afrag(A0, A1, A2, A3);
        #pragma unroll
        for (int t = 0; t < 8; ++t) {
            acc[t] = __builtin_amdgcn_wmma_f32_16x16x32_bf16(
                false, a, false, B0[t], (short)0, acc[t], false, false);
        }

        if (kk < 7) {
            A0 = N0; A1 = N1; A2 = N2; A3 = N3;
            #pragma unroll
            for (int t = 0; t < 8; ++t) B0[t] = B1[t];
        }
    }

    // D layout: VGPR r, lanes 0-15 -> M=r, lanes 16-31 -> M=r+8; N = m
    #pragma unroll
    for (int t = 0; t < 8; ++t) {
        int col0 = cg * 128 + t * 16;
        #pragma unroll
        for (int r = 0; r < 8; ++r) {
            xt[(size_t)(row0 + r + 8 * h) * OUTF + col0 + m] = acc[t][r];
        }
    }
}

// ---------------- per-node attention logits: a_src/a_dst [N,4] ----------------
__global__ void k_logits(const float* __restrict__ xt, const float* __restrict__ att_src,
                         const float* __restrict__ att_dst, float* __restrict__ asrc,
                         float* __restrict__ adst, int N) {
    int idx = blockIdx.x * blockDim.x + threadIdx.x;   // n*4 + h
    if (idx >= N * HEADS) return;
    int hh = idx & 3, n = idx >> 2;
    const float4* row = (const float4*)(xt + (size_t)n * OUTF + hh * HID);
    const float4* a1  = (const float4*)(att_src + hh * HID);
    const float4* a2  = (const float4*)(att_dst + hh * HID);
    float ss = 0.f, sd = 0.f;
    #pragma unroll
    for (int d = 0; d < HID / 4; ++d) {
        float4 v = row[d], w1 = a1[d], w2 = a2[d];
        ss += v.x * w1.x + v.y * w1.y + v.z * w1.z + v.w * w1.w;
        sd += v.x * w2.x + v.y * w2.y + v.z * w2.z + v.w * w2.w;
    }
    asrc[idx] = ss;
    adst[idx] = sd;
}

// ---------------- softmax denominators: denom[dst][h] += exp(leaky(e)) ----------------
__global__ void k_denom(const int* __restrict__ ei, const float* __restrict__ asrc,
                        const float* __restrict__ adst, float* __restrict__ denom, int E) {
    int e = blockIdx.x * blockDim.x + threadIdx.x;
    if (e >= E) return;
    int s = ei[e], d = ei[E + e];
    float4 as = *(const float4*)(asrc + (size_t)s * 4);
    float4 ad = *(const float4*)(adst + (size_t)d * 4);
    float e0 = as.x + ad.x; e0 = e0 > 0.f ? e0 : NEG_SLOPE * e0;
    float e1 = as.y + ad.y; e1 = e1 > 0.f ? e1 : NEG_SLOPE * e1;
    float e2 = as.z + ad.z; e2 = e2 > 0.f ? e2 : NEG_SLOPE * e2;
    float e3 = as.w + ad.w; e3 = e3 > 0.f ? e3 : NEG_SLOPE * e3;
    float* dn = denom + (size_t)d * 4;
    atomicAdd(dn + 0, __expf(e0));
    atomicAdd(dn + 1, __expf(e1));
    atomicAdd(dn + 2, __expf(e2));
    atomicAdd(dn + 3, __expf(e3));
}

// ---------------- pooled-sum aggregation: psum[f] = sum_e alpha[e,h(f)]*xt[src_e,f] ----
// Block stages 64 edges' alphas + src ids in LDS; thread tid owns feature tid and
// accumulates in a register; one global atomicAdd per thread at the end.
__global__ void k_agg(const int* __restrict__ ei, const float* __restrict__ asrc,
                      const float* __restrict__ adst, const float* __restrict__ denom,
                      const float* __restrict__ xt, float* __restrict__ psum,
                      int E, int nBatches) {
    __shared__ float s_alpha[64 * 4];
    __shared__ int   s_src[64];
    int tid  = threadIdx.x;
    int head = tid >> 6;
    float acc = 0.f;
    for (int b = blockIdx.x; b < nBatches; b += gridDim.x) {
        int e  = b * 64 + (tid >> 2);
        int hh = tid & 3;
        float al = 0.f;
        int s = 0;
        if (e < E) {
            s = ei[e];
            int d = ei[E + e];
            float ev = asrc[(size_t)s * 4 + hh] + adst[(size_t)d * 4 + hh];
            ev = ev > 0.f ? ev : NEG_SLOPE * ev;
            al = __expf(ev) / (denom[(size_t)d * 4 + hh] + 1e-16f);
        }
        s_alpha[tid] = al;                // == s_alpha[(tid>>2)*4 + hh]
        if (hh == 0) s_src[tid >> 2] = s;
        __syncthreads();
        #pragma unroll 4
        for (int j = 0; j < 64; ++j)
            acc += s_alpha[j * 4 + head] * xt[(size_t)s_src[j] * OUTF + tid];
        __syncthreads();
    }
    atomicAdd(&psum[tid], acc);
}

// ---------------- head: mean-pool + bias, MLP, double sigmoid ----------------
__global__ void k_head(const float* __restrict__ psum, const float* __restrict__ bias,
                       const float* __restrict__ W1, const float* __restrict__ b1,
                       const float* __restrict__ W2, const float* __restrict__ b2,
                       float* __restrict__ out, float invN) {
    __shared__ float p[OUTF];
    __shared__ float t[HID];
    int tid = threadIdx.x;
    p[tid] = psum[tid] * invN + bias[tid];
    __syncthreads();
    if (tid < HID) {
        float s = b1[tid];
        #pragma unroll 4
        for (int f = 0; f < OUTF; ++f) s += p[f] * W1[f * HID + tid];
        s = fmaxf(s, 0.f);
        t[tid] = s * W2[tid];
    }
    __syncthreads();
    if (tid == 0) {
        float s = b2[0];
        for (int j = 0; j < HID; ++j) s += t[j];
        float logit = 1.f / (1.f + __expf(-s));
        out[0] = 1.f / (1.f + __expf(-logit));    // reference applies sigmoid twice
    }
}

extern "C" void kernel_launch(void* const* d_in, const int* in_sizes, int n_in,
                              void* d_out, int out_size, void* d_ws, size_t ws_size,
                              hipStream_t stream) {
    (void)n_in; (void)out_size; (void)ws_size;
    const float* x       = (const float*)d_in[0];
    const int*   ei      = (const int*)d_in[1];   // edge_index, [2,E] flat
    const float* W       = (const float*)d_in[3];
    const float* att_src = (const float*)d_in[4];
    const float* att_dst = (const float*)d_in[5];
    const float* bias    = (const float*)d_in[6];
    const float* W1      = (const float*)d_in[7];
    const float* b1      = (const float*)d_in[8];
    const float* W2      = (const float*)d_in[9];
    const float* b2      = (const float*)d_in[10];
    float* out = (float*)d_out;

    const int N = in_sizes[0] / IN_DIM;   // 50000
    const int E = in_sizes[1] / 2;        // 800000

    // workspace layout
    char* ws = (char*)d_ws;
    bf16_t* Wt = (bf16_t*)ws;                        size_t off = (size_t)IN_DIM * OUTF * 2;
    float* xt    = (float*)(ws + off);               off += (size_t)N * OUTF * 4;
    float* asrc  = (float*)(ws + off);               off += (size_t)N * HEADS * 4;
    float* adst  = (float*)(ws + off);               off += (size_t)N * HEADS * 4;
    float* denom = (float*)(ws + off);               off += (size_t)N * HEADS * 4;
    float* psum  = (float*)(ws + off);               // 256 floats; contiguous with denom

    // 1) zero accumulators (denom + psum are contiguous)
    {
        int zn = N * HEADS + OUTF;
        k_zero<<<(zn + 255) / 256, 256, 0, stream>>>(denom, zn);
    }
    // 2) W -> Wt bf16
    k_wt<<<(IN_DIM * OUTF + 255) / 256, 256, 0, stream>>>(W, Wt);
    // 3) GEMM (WMMA bf16, double-buffered K loop)
    {
        int Mtiles = N / 16;
        k_gemm<<<(Mtiles + 3) / 4, 256, 0, stream>>>(x, Wt, xt, Mtiles);
    }
    // 4) per-node logits
    k_logits<<<(N * HEADS + 255) / 256, 256, 0, stream>>>(xt, att_src, att_dst, asrc, adst, N);
    // 5) softmax denominators
    k_denom<<<(E + 255) / 256, 256, 0, stream>>>(ei, asrc, adst, denom, E);
    // 6) pooled-sum aggregation
    {
        int nBatches = (E + 63) / 64;
        int grid = nBatches < 1024 ? nBatches : 1024;
        k_agg<<<grid, 256, 0, stream>>>(ei, asrc, adst, denom, xt, psum, E, nBatches);
    }
    // 7) head MLP -> scalar
    k_head<<<1, 256, 0, stream>>>(psum, bias, W1, b1, W2, b2, out, 1.0f / (float)N);
}
